// MambaBlock_8495445311703
// MI455X (gfx1250) — matrix operand order repaired and checked by reading, not compiled, not run on hardware
//
#include <hip/hip_runtime.h>

// ---------------------------------------------------------------------------
// Feature detection for CDNA5 async-copy-to-LDS path
// ---------------------------------------------------------------------------
#if defined(__has_builtin)
#if __has_builtin(__builtin_amdgcn_global_load_async_to_lds_b128)
#define ASYNC_LDS_AVAILABLE 1
#else
#define ASYNC_LDS_AVAILABLE 0
#endif
#else
#define ASYNC_LDS_AVAILABLE 0
#endif

template <int N>
__device__ __forceinline__ void wait_asynccnt() {
#if defined(__has_builtin) && __has_builtin(__builtin_amdgcn_s_wait_asynccnt)
    __builtin_amdgcn_s_wait_asynccnt(N);
#else
    asm volatile("s_wait_asynccnt %0" ::"i"(N));
#endif
}

// ---------------------------------------------------------------------------
// Types for CDNA5 WMMA / async copies
// ---------------------------------------------------------------------------
typedef __attribute__((ext_vector_type(16))) __bf16       v16bf;
typedef __attribute__((ext_vector_type(8)))  float        v8f;
typedef __attribute__((ext_vector_type(4)))  unsigned int uint4v;
typedef __attribute__((ext_vector_type(4)))  int          int4v;

// AS-qualified pointee types for the async-copy builtin
typedef __attribute__((address_space(1))) int4v as1_int4v;
typedef __attribute__((address_space(3))) int4v as3_int4v;

union FragBF {
    uint4v u[2];
    v16bf  v;
};
static_assert(sizeof(FragBF) == 32, "frag size");

// ---------------------------------------------------------------------------
// Scalar helpers
// ---------------------------------------------------------------------------
__device__ __forceinline__ unsigned short f2bf(float f) {
    unsigned int u = __float_as_uint(f);
    return (unsigned short)((u + 0x7FFFu + ((u >> 16) & 1u)) >> 16);
}
__device__ __forceinline__ float bf2f(unsigned short h) {
    return __uint_as_float(((unsigned int)h) << 16);
}
__device__ __forceinline__ float silu_f(float x) {
    return x / (1.0f + __expf(-x));
}
__device__ __forceinline__ float softplus_f(float x) {
    return (x > 20.0f) ? x : __logf(1.0f + __expf(x));
}

// ---------------------------------------------------------------------------
// f32 -> bf16 conversion (for weights)
// ---------------------------------------------------------------------------
__global__ void __launch_bounds__(256) k_f32_to_bf16(const float* __restrict__ src,
                                                     unsigned short* __restrict__ dst,
                                                     int n) {
    int i = blockIdx.x * blockDim.x + threadIdx.x;
    if (i < n) dst[i] = f2bf(src[i]);
}

// ---------------------------------------------------------------------------
// LayerNorm over D=1024, one block (256 threads) per row, bf16 output
// ---------------------------------------------------------------------------
__global__ void __launch_bounds__(256) k_layernorm_bf16(const float* __restrict__ x,
                                                        const float* __restrict__ g,
                                                        const float* __restrict__ b,
                                                        unsigned short* __restrict__ out) {
    __shared__ float red[256];
    const int row = blockIdx.x;
    const int tid = threadIdx.x;
    const float* xr = x + (size_t)row * 1024;
    float v0 = xr[tid], v1 = xr[tid + 256], v2 = xr[tid + 512], v3 = xr[tid + 768];
    red[tid] = v0 + v1 + v2 + v3;
    __syncthreads();
    for (int o = 128; o > 0; o >>= 1) {
        if (tid < o) red[tid] += red[tid + o];
        __syncthreads();
    }
    const float mu = red[0] * (1.0f / 1024.0f);
    __syncthreads();
    float d0 = v0 - mu, d1 = v1 - mu, d2 = v2 - mu, d3 = v3 - mu;
    red[tid] = d0 * d0 + d1 * d1 + d2 * d2 + d3 * d3;
    __syncthreads();
    for (int o = 128; o > 0; o >>= 1) {
        if (tid < o) red[tid] += red[tid + o];
        __syncthreads();
    }
    const float rstd = rsqrtf(red[0] * (1.0f / 1024.0f) + 1e-5f);
    unsigned short* orow = out + (size_t)row * 1024;
    orow[tid]       = f2bf(d0 * rstd * g[tid]       + b[tid]);
    orow[tid + 256] = f2bf(d1 * rstd * g[tid + 256] + b[tid + 256]);
    orow[tid + 512] = f2bf(d2 * rstd * g[tid + 512] + b[tid + 512]);
    orow[tid + 768] = f2bf(d3 * rstd * g[tid + 768] + b[tid + 768]);
}

// ---------------------------------------------------------------------------
// Tiled WMMA GEMM:   out[M,N] = epilogue( A[M,K] * W[N,K]^T )
// A, W are bf16 row-major. Block tile = (WROWS*64) x (WCOLS*32), BK = 32.
// Each wave computes 64x32 via 4x2 fragments of v_wmma_f32_16x16x32_bf16.
// STAGE=true (requires 2x4 waves / 256 threads): double-buffered LDS staging
// with global_load_async_to_lds_b128 + s_wait_asynccnt; fragments come from
// LDS (ds_load_b128). STAGE=false: direct global fragment loads.
// Epilogues: 0=f32, 1=bf16, 2=bias+silu->bf16, 3=(+bias)+residual->f32,
//            4=softplus(+bias)->f32
// ---------------------------------------------------------------------------
template <int EPI, int WROWS, int WCOLS, bool STAGE>
__global__ void __launch_bounds__(WROWS * WCOLS * 32)
k_gemm_bf16_wmma(const unsigned short* __restrict__ A,
                 const unsigned short* __restrict__ W,
                 const float* __restrict__ bias,
                 const float* __restrict__ residual,
                 float* __restrict__ outF,
                 unsigned short* __restrict__ outB,
                 int M, int N, int K) {
    constexpr int FM = 4, FN = 2;
    constexpr int BM = WROWS * FM * 16;   // 128
    constexpr int BN = WCOLS * FN * 16;   // 128 (or 96)
    constexpr bool kStage = STAGE && (ASYNC_LDS_AVAILABLE != 0);

    const int lane  = threadIdx.x & 31;
    const int wave  = threadIdx.x >> 5;
    const int wn    = wave % WCOLS;
    const int wm    = wave / WCOLS;
    const int blockM  = blockIdx.y * BM;
    const int blockN  = blockIdx.x * BN;
    const int rowBase = blockM + wm * FM * 16;
    const int colBase = blockN + wn * FN * 16;
    const int mlane = lane & 15;
    const int half  = lane >> 4;

    v8f acc[FM][FN];
#pragma unroll
    for (int i = 0; i < FM; i++)
#pragma unroll
        for (int j = 0; j < FN; j++)
#pragma unroll
            for (int e = 0; e < 8; e++) acc[i][j][e] = 0.0f;

#if ASYNC_LDS_AVAILABLE
    if constexpr (kStage) {
        static_assert(!STAGE || (WROWS == 2 && WCOLS == 4), "staged path needs 256 threads");
        // double-buffered LDS tiles: [buf][row][32] bf16
        __shared__ unsigned short sA[2][BM * 32];
        __shared__ unsigned short sB[2][BN * 32];
        const int t = threadIdx.x;

        // Each thread async-copies 2 chunks of A and 2 of B (16B each) per buffer.
        auto issue = [&](int buf, int k) {
#pragma unroll
            for (int q = 0; q < 2; q++) {
                const int c   = t * 2 + q;       // 0..511
                const int row = c >> 2;          // 0..127
                const int cc  = c & 3;           // 16B chunk within 64B row
                const unsigned short* gpA = A + (size_t)(blockM + row) * K + k + cc * 8;
                unsigned short*       lpA = &sA[buf][row * 32 + cc * 8];
                __builtin_amdgcn_global_load_async_to_lds_b128(
                    (as1_int4v*)gpA, (as3_int4v*)lpA, 0, 0);
                const unsigned short* gpB = W + (size_t)(blockN + row) * K + k + cc * 8;
                unsigned short*       lpB = &sB[buf][row * 32 + cc * 8];
                __builtin_amdgcn_global_load_async_to_lds_b128(
                    (as1_int4v*)gpB, (as3_int4v*)lpB, 0, 0);
            }
        };

        const int nk = K / 32;
        issue(0, 0);
        for (int ki = 0; ki < nk; ki++) {
            const int cur = ki & 1;
            if (ki + 1 < nk) {
                issue(1 - cur, (ki + 1) * 32);
                wait_asynccnt<4>();   // 4 copies of next buffer may remain in flight
            } else {
                wait_asynccnt<0>();
            }
            __syncthreads();

            FragBF a[FM], b[FN];
#pragma unroll
            for (int i = 0; i < FM; i++) {
                const unsigned short* p =
                    &sA[cur][(wm * 64 + i * 16 + mlane) * 32 + half * 8];
                a[i].u[0] = *(const uint4v*)(p);
                a[i].u[1] = *(const uint4v*)(p + 16);
            }
#pragma unroll
            for (int j = 0; j < FN; j++) {
                const unsigned short* p =
                    &sB[cur][(wn * 32 + j * 16 + mlane) * 32 + half * 8];
                b[j].u[0] = *(const uint4v*)(p);
                b[j].u[1] = *(const uint4v*)(p + 16);
            }
#pragma unroll
            for (int i = 0; i < FM; i++)
#pragma unroll
                for (int j = 0; j < FN; j++)
                    acc[i][j] = __builtin_amdgcn_wmma_f32_16x16x32_bf16(
                        false, a[i].v, false, b[j].v, (short)0, acc[i][j], false, false);

            __syncthreads();  // don't overwrite cur buffer until everyone consumed it
        }
    }
#endif
    if constexpr (!kStage) {
        for (int k = 0; k < K; k += 32) {
            FragBF a[FM], b[FN];
#pragma unroll
            for (int i = 0; i < FM; i++) {
                const unsigned short* p =
                    A + (size_t)(rowBase + i * 16 + mlane) * K + k + half * 8;
                a[i].u[0] = *(const uint4v*)(p);
                a[i].u[1] = *(const uint4v*)(p + 16);
            }
#pragma unroll
            for (int j = 0; j < FN; j++) {
                const unsigned short* p =
                    W + (size_t)(colBase + j * 16 + mlane) * K + k + half * 8;
                b[j].u[0] = *(const uint4v*)(p);
                b[j].u[1] = *(const uint4v*)(p + 16);
            }
#pragma unroll
            for (int i = 0; i < FM; i++)
#pragma unroll
                for (int j = 0; j < FN; j++)
                    acc[i][j] = __builtin_amdgcn_wmma_f32_16x16x32_bf16(
                        false, a[i].v, false, b[j].v, (short)0, acc[i][j], false, false);
        }
    }

    // Epilogue: C/D layout — lanes 0-15 hold M=e, lanes 16-31 hold M=8+e, N=lane%16
#pragma unroll
    for (int i = 0; i < FM; i++) {
#pragma unroll
        for (int j = 0; j < FN; j++) {
#pragma unroll
            for (int e = 0; e < 8; e++) {
                const int row = rowBase + i * 16 + half * 8 + e;
                const int col = colBase + j * 16 + mlane;
                const size_t idx = (size_t)row * N + col;
                float v = acc[i][j][e];
                if (EPI == 0) {
                    outF[idx] = v;
                } else if (EPI == 1) {
                    outB[idx] = f2bf(v);
                } else if (EPI == 2) {
                    v += bias[col];
                    outB[idx] = f2bf(silu_f(v));
                } else if (EPI == 3) {
                    if (bias) v += bias[col];
                    outF[idx] = v + residual[idx];
                } else {
                    v += bias[col];
                    outF[idx] = softplus_f(v);
                }
            }
        }
    }
}

// ---------------------------------------------------------------------------
// Depthwise causal conv1d (k=4) + SiLU over u = xz[:, 0:4096]  -> bf16
// ---------------------------------------------------------------------------
__global__ void __launch_bounds__(256) k_conv_silu(const unsigned short* __restrict__ xz,
                                                   const float* __restrict__ cw,
                                                   const float* __restrict__ cb,
                                                   unsigned short* __restrict__ uconv,
                                                   int total) {
    int i = blockIdx.x * blockDim.x + threadIdx.x;
    if (i >= total) return;
    const int d  = i & 4095;
    const int t  = (i >> 12) & 2047;
    const int bb = i >> 23;
    __builtin_prefetch(xz + (size_t)(bb * 2048 + t) * 8192 + d, 0, 1);
    float acc = cb[d];
#pragma unroll
    for (int kk = 0; kk < 4; kk++) {
        const int tt = t - 3 + kk;
        if (tt >= 0)
            acc += bf2f(xz[(size_t)(bb * 2048 + tt) * 8192 + d]) * cw[d * 4 + kk];
    }
    uconv[i] = f2bf(silu_f(acc));
}

// ---------------------------------------------------------------------------
// Split x_dbl[4096,96] -> dt_in (bf16 [.,64]), Bm (f32 [.,16]), Cm (f32 [.,16])
// ---------------------------------------------------------------------------
__global__ void __launch_bounds__(256) k_split_xdbl(const float* __restrict__ xdbl,
                                                    unsigned short* __restrict__ dtin,
                                                    float* __restrict__ Bm,
                                                    float* __restrict__ Cm,
                                                    int rows) {
    int i = blockIdx.x * blockDim.x + threadIdx.x;
    if (i >= rows * 96) return;
    const int r = i / 96, c = i % 96;
    const float v = xdbl[i];
    if (c < 64)      dtin[(size_t)r * 64 + c] = f2bf(v);
    else if (c < 80) Bm[(size_t)r * 16 + (c - 64)] = v;
    else             Cm[(size_t)r * 16 + (c - 80)] = v;
}

// ---------------------------------------------------------------------------
// Selective scan: one lane per channel; h[16] state in VGPRs; B/C via LDS.
// Fused epilogue: y = (scan + u*D) * silu(z)  -> bf16
// ---------------------------------------------------------------------------
__global__ void __launch_bounds__(256) k_selective_scan(const float* __restrict__ dt,
                                                        const unsigned short* __restrict__ uconv,
                                                        const unsigned short* __restrict__ xz,
                                                        const float* __restrict__ Bm,
                                                        const float* __restrict__ Cm,
                                                        const float* __restrict__ A_log,
                                                        const float* __restrict__ Dp,
                                                        unsigned short* __restrict__ y) {
    __shared__ float sB[8 * 16];
    __shared__ float sC[8 * 16];
    const int d  = blockIdx.y * 256 + threadIdx.x;
    const int bb = blockIdx.x;
    float A[16], h[16];
#pragma unroll
    for (int s = 0; s < 16; s++) {
        A[s] = -__expf(A_log[(size_t)d * 16 + s]);
        h[s] = 0.0f;
    }
    const float Dd = Dp[d];
    for (int t0 = 0; t0 < 2048; t0 += 8) {
        const int tid = threadIdx.x;
        if (tid < 128) {
            const int tt = tid >> 4, s = tid & 15;
            sB[tid] = Bm[(size_t)(bb * 2048 + t0 + tt) * 16 + s];
        } else {
            const int q = tid - 128;
            const int tt = q >> 4, s = q & 15;
            sC[q] = Cm[(size_t)(bb * 2048 + t0 + tt) * 16 + s];
        }
        __syncthreads();
#pragma unroll
        for (int tt = 0; tt < 8; tt++) {
            const size_t row = (size_t)(bb * 2048 + t0 + tt);
            const float dtv = dt[row * 4096 + d];
            const float uv  = bf2f(uconv[row * 4096 + d]);
            const float du  = dtv * uv;
            float yv = 0.0f;
#pragma unroll
            for (int s = 0; s < 16; s++) {
                const float dA = __expf(dtv * A[s]);
                h[s] = dA * h[s] + du * sB[tt * 16 + s];
                yv += h[s] * sC[tt * 16 + s];
            }
            const float zv = bf2f(xz[row * 8192 + 4096 + d]);
            yv = (yv + uv * Dd) * silu_f(zv);
            y[row * 4096 + d] = f2bf(yv);
        }
        __syncthreads();
    }
}

// ---------------------------------------------------------------------------
// Host orchestration
// ---------------------------------------------------------------------------
extern "C" void kernel_launch(void* const* d_in, const int* in_sizes, int n_in,
                              void* d_out, int out_size, void* d_ws, size_t ws_size,
                              hipStream_t stream) {
    (void)in_sizes; (void)n_in; (void)out_size; (void)ws_size;

    // ---- inputs (setup_inputs order, all f32) ----
    const float* x        = (const float*)d_in[0];
    const float* g1       = (const float*)d_in[1];
    const float* b1       = (const float*)d_in[2];
    const float* g2       = (const float*)d_in[3];
    const float* b2       = (const float*)d_in[4];
    const float* w1       = (const float*)d_in[5];
    const float* bias1    = (const float*)d_in[6];
    const float* w2       = (const float*)d_in[7];
    const float* bias2    = (const float*)d_in[8];
    const float* in_projw = (const float*)d_in[9];
    const float* conv_w   = (const float*)d_in[10];
    const float* conv_b   = (const float*)d_in[11];
    const float* x_projw  = (const float*)d_in[12];
    const float* dt_projw = (const float*)d_in[13];
    const float* dt_projb = (const float*)d_in[14];
    const float* A_log    = (const float*)d_in[15];
    const float* Dvec     = (const float*)d_in[16];
    const float* out_projw= (const float*)d_in[17];

    const int MT = 4096;  // B*L tokens

    // ---- workspace carve-out ----
    size_t off = 0;
    auto alloc = [&](size_t bytes) -> void* {
        void* p = (char*)d_ws + off;
        off += (bytes + 255) & ~(size_t)255;
        return p;
    };
    unsigned short* w1b    = (unsigned short*)alloc((size_t)4096 * 1024 * 2);
    unsigned short* w2b    = (unsigned short*)alloc((size_t)1024 * 4096 * 2);
    unsigned short* inpb   = (unsigned short*)alloc((size_t)8192 * 1024 * 2);
    unsigned short* xpb    = (unsigned short*)alloc((size_t)96 * 4096 * 2);
    unsigned short* dtpb   = (unsigned short*)alloc((size_t)4096 * 64 * 2);
    unsigned short* opb    = (unsigned short*)alloc((size_t)1024 * 4096 * 2);
    unsigned short* hb     = (unsigned short*)alloc((size_t)MT * 1024 * 2);   // ln1 out
    unsigned short* h2b    = (unsigned short*)alloc((size_t)MT * 4096 * 2);   // mlp mid
    float*          x1f    = (float*)alloc((size_t)MT * 1024 * 4);            // residual 1
    unsigned short* xnb    = (unsigned short*)alloc((size_t)MT * 1024 * 2);   // ln2 out
    unsigned short* xzb    = (unsigned short*)alloc((size_t)MT * 8192 * 2);   // u|z
    unsigned short* uconvb = (unsigned short*)alloc((size_t)MT * 4096 * 2);
    float*          xdbl   = (float*)alloc((size_t)MT * 96 * 4);
    unsigned short* dtinb  = (unsigned short*)alloc((size_t)MT * 64 * 2);
    float*          Bmf    = (float*)alloc((size_t)MT * 16 * 4);
    float*          Cmf    = (float*)alloc((size_t)MT * 16 * 4);
    float*          dtf    = (float*)alloc((size_t)MT * 4096 * 4);
    unsigned short* yb     = (unsigned short*)alloc((size_t)MT * 4096 * 2);

    // ---- weight conversion ----
    auto cvt = [&](const float* s, unsigned short* d, int n) {
        k_f32_to_bf16<<<(n + 255) / 256, 256, 0, stream>>>(s, d, n);
    };
    cvt(w1, w1b, 4096 * 1024);
    cvt(w2, w2b, 1024 * 4096);
    cvt(in_projw, inpb, 8192 * 1024);
    cvt(x_projw, xpb, 96 * 4096);
    cvt(dt_projw, dtpb, 4096 * 64);
    cvt(out_projw, opb, 1024 * 4096);

    // ---- LN1 ----
    k_layernorm_bf16<<<MT, 256, 0, stream>>>(x, g1, b1, hb);

    // ---- MLP up: silu(h @ w1^T + bias1) -> bf16 [4096,4096] ----
    k_gemm_bf16_wmma<2, 2, 4, true><<<dim3(32, 32), 256, 0, stream>>>(
        hb, w1b, bias1, nullptr, nullptr, h2b, MT, 4096, 1024);

    // ---- MLP down + residual: x1 = x + h2 @ w2^T + bias2 ----
    k_gemm_bf16_wmma<3, 2, 4, true><<<dim3(8, 32), 256, 0, stream>>>(
        h2b, w2b, bias2, x, x1f, nullptr, MT, 1024, 4096);

    // ---- LN2 ----
    k_layernorm_bf16<<<MT, 256, 0, stream>>>(x1f, g2, b2, xnb);

    // ---- in_proj: xz = xn @ in_proj_w^T -> bf16 [4096,8192] ----
    k_gemm_bf16_wmma<1, 2, 4, true><<<dim3(64, 32), 256, 0, stream>>>(
        xnb, inpb, nullptr, nullptr, nullptr, xzb, MT, 8192, 1024);

    // ---- causal depthwise conv + silu ----
    k_conv_silu<<<(MT * 4096) / 256, 256, 0, stream>>>(xzb, conv_w, conv_b, uconvb, MT * 4096);

    // ---- x_proj: x_dbl = u @ x_proj_w^T -> f32 [4096,96] (N=96 tile, 6 waves) ----
    k_gemm_bf16_wmma<0, 2, 3, false><<<dim3(1, 32), 192, 0, stream>>>(
        uconvb, xpb, nullptr, nullptr, xdbl, nullptr, MT, 96, 4096);

    // ---- split into dt_in / B / C ----
    k_split_xdbl<<<(MT * 96 + 255) / 256, 256, 0, stream>>>(xdbl, dtinb, Bmf, Cmf, MT);

    // ---- dt: softplus(dt_in @ dt_proj_w^T + dt_proj_b) -> f32 [4096,4096] ----
    k_gemm_bf16_wmma<4, 2, 4, true><<<dim3(32, 32), 256, 0, stream>>>(
        dtinb, dtpb, dt_projb, nullptr, dtf, nullptr, MT, 4096, 64);

    // ---- selective scan (fused +u*D and *silu(z)) -> bf16 y ----
    k_selective_scan<<<dim3(2, 16), 256, 0, stream>>>(
        dtf, uconvb, xzb, Bmf, Cmf, A_log, Dvec, yb);

    // ---- out_proj + residual: out = x1 + y @ out_proj_w^T ----
    k_gemm_bf16_wmma<3, 2, 4, true><<<dim3(8, 32), 256, 0, stream>>>(
        yb, opb, nullptr, x1f, (float*)d_out, nullptr, MT, 1024, 4096);
}
